// TSK_FS_8435315769752
// MI455X (gfx1250) — matrix operand order repaired and checked by reading, not compiled, not run on hardware
//
#include <hip/hip_runtime.h>
#include <hip/hip_bf16.h>

// TSK fuzzy system, restructured:
//   out[n] = ( Gb[n] + sum_m x[n,m]*G[n,m] ) / prod_m ( sum_d mg[n,m,d] )
// where [G|Gb] = fg @ [W|b]  (fg generated on the fly from base-4 digit products)
// GEMM: (1024 x 65536) x (65536 x 9-pad-16), via v_wmma_f32_16x16x32_bf16.
//
// d_ws layout:
//   [0, 2MB)      : Bpack — [W|b|0] pre-swizzled into per-lane WMMA B-fragment
//                   layout: Bpack[chunk(2048)][lane(32)][16 x bf16] (32B/lane)
//   [2MB, 4MB)    : f32 partial C tiles: [slice(32)][tile(64)][16][16]

#define MDIM 8
#define NMF  4
#define R_RULES 65536
#define N_SAMPLES 1024
#define NTILES (N_SAMPLES / 16)      // 64 row-tiles of 16 samples
#define BPT 4                        // blocks per tile
#define WPB 8                        // waves per block (256 threads, wave32)
#define SLICES (BPT * WPB)           // 32 K-slices per tile
#define KCHUNK 32                    // K per WMMA
#define NCHUNKS (R_RULES / KCHUNK)   // 2048
#define CHUNKS_PER_WAVE (NCHUNKS / SLICES)  // 64
#define BPACK_U32 (NCHUNKS * 32 * 8) // 512K u32 words = 2MB

typedef __attribute__((ext_vector_type(16))) __bf16       v16bf;
typedef __attribute__((ext_vector_type(8)))  float        v8f;
typedef __attribute__((ext_vector_type(8)))  unsigned int v8u;

// ---------------------------------------------------------------------------
// Prep: swizzle [W|b|0] (f32) into bf16 B-fragment layout.
// Word w -> chunk c = w>>8; lane l = (w>>3)&31; pair j = w&7 (elements 2j,2j+1).
// Element t of lane (lo,hi) holds Wb[r = c*32 + hi*16 + t][col = lo].
// ---------------------------------------------------------------------------
__global__ __launch_bounds__(256) void tsk_pack_b(
    const float* __restrict__ W, const float* __restrict__ b,
    unsigned int* __restrict__ bpack)
{
    const int w = blockIdx.x * 256 + threadIdx.x;   // 0 .. BPACK_U32-1
    const int c   = w >> 8;
    const int rem = w & 255;
    const int l   = rem >> 3;
    const int j   = rem & 7;
    const int lo = l & 15, hi = l >> 4;
    const int rr = c * KCHUNK + hi * 16 + j * 2;

    float v0, v1;
    if (lo < 8)       { v0 = W[(size_t)rr * MDIM + lo]; v1 = W[(size_t)(rr + 1) * MDIM + lo]; }
    else if (lo == 8) { v0 = b[rr];                     v1 = b[rr + 1]; }
    else              { v0 = 0.0f;                      v1 = 0.0f; }

    const unsigned short u0 = __builtin_bit_cast(unsigned short, (__bf16)v0);
    const unsigned short u1 = __builtin_bit_cast(unsigned short, (__bf16)v1);
    bpack[w] = (unsigned int)u0 | ((unsigned int)u1 << 16);
}

// ---------------------------------------------------------------------------
// A-fragment (fg) builder: row = lo, element e -> Kc = hi*16 + e.
// Top-5 digit prefix shared per chunk; each 8-run shares d5 and d6-base.
// ---------------------------------------------------------------------------
__device__ __forceinline__ v16bf make_afrag(const float (*mgr)[NMF], int r0,
                                            int hi, float pre04)
{
    v16bf afrag;
    #pragma unroll
    for (int runi = 0; runi < 2; ++runi) {
        const int rr = r0 + hi * 16 + runi * 8;        // multiple of 8
        const float m5   = mgr[5][(rr >> 4) & 3];
        const int   d6b  = (rr >> 2) & 3;              // even
        const float preA = pre04 * m5 * mgr[6][d6b];
        const float preB = pre04 * m5 * mgr[6][d6b + 1];
        #pragma unroll
        for (int t = 0; t < 8; ++t) {
            const float fg = ((t < 4) ? preA : preB) * mgr[7][t & 3];
            afrag[runi * 8 + t] = (__bf16)fg;
        }
    }
    return afrag;
}

__device__ __forceinline__ float prefix04(const float (*mgr)[NMF], int r0)
{
    return mgr[0][(r0 >> 14) & 3] * mgr[1][(r0 >> 12) & 3] *
           mgr[2][(r0 >> 10) & 3] * mgr[3][(r0 >>  8) & 3] *
           mgr[4][(r0 >>  6) & 3];
}

// ---------------------------------------------------------------------------
// Main GEMM: each wave owns one (tile, slice); 64 chunks, 2x unrolled with
// two independent accumulators to break the WMMA dependency chain.
// ---------------------------------------------------------------------------
__global__ __launch_bounds__(256) void tsk_wmma_main(
    const float* __restrict__ x,      // (1024, 8)
    const float* __restrict__ center, // (8, 4)
    const float* __restrict__ sigma,  // (8, 4)
    const v8u*   __restrict__ bpack,  // (2048, 32) 32B B-fragment slabs
    float*       __restrict__ part)   // (32, 64, 16, 16) f32 partials
{
    const int tile      = blockIdx.x / BPT;
    const int blkInTile = blockIdx.x % BPT;
    const int wave      = threadIdx.x >> 5;
    const int lane      = threadIdx.x & 31;
    const int slice     = blkInTile * WPB + wave;  // 0..31

    __shared__ float mg[16][MDIM][NMF];            // 2 KB
    for (int i = threadIdx.x; i < 16 * MDIM * NMF; i += 256) {
        const int row = i >> 5;
        const int m   = (i >> 2) & 7;
        const int d   = i & 3;
        const int n   = tile * 16 + row;
        const float diff = x[n * MDIM + m] - center[m * NMF + d];
        const float sg   = sigma[m * NMF + d];
        mg[row][m][d] = __expf(-diff * diff * 0.5f * sg * sg);
    }
    __syncthreads();

    const int lo = lane & 15;
    const int hi = lane >> 4;
    const float (*mgr)[NMF] = mg[lo];

    v8f acc0 = {}, acc1 = {};

    #pragma unroll 1
    for (int j = 0; j < CHUNKS_PER_WAVE; j += 2) {
        const int chunk0 = slice + j * SLICES;
        const int chunk1 = chunk0 + SLICES;

        // Speculative prefetch of the B-fragments two chunks ahead
        // (past-the-end addresses are silently dropped).
        __builtin_prefetch(&bpack[(size_t)(chunk0 + 2 * SLICES) * 32 + lane], 0, 1);

        const v16bf b0 = __builtin_bit_cast(v16bf, bpack[(size_t)chunk0 * 32 + lane]);
        const v16bf b1 = __builtin_bit_cast(v16bf, bpack[(size_t)chunk1 * 32 + lane]);

        const int r0a = chunk0 * KCHUNK;
        const int r0b = chunk1 * KCHUNK;
        const v16bf a0 = make_afrag(mgr, r0a, hi, prefix04(mgr, r0a));
        const v16bf a1 = make_afrag(mgr, r0b, hi, prefix04(mgr, r0b));

        acc0 = __builtin_amdgcn_wmma_f32_16x16x32_bf16(
            false, a0, false, b0, (short)0, acc0, false, false);
        acc1 = __builtin_amdgcn_wmma_f32_16x16x32_bf16(
            false, a1, false, b1, (short)0, acc1, false, false);
    }

    // C layout: element v sits at (row = v + 8*hi, col = lo).
    float* wsp = part + ((size_t)slice * NTILES + tile) * 256;
    #pragma unroll
    for (int v = 0; v < 8; ++v)
        wsp[(v + 8 * hi) * 16 + lo] = acc0[v] + acc1[v];
}

// ---------------------------------------------------------------------------
// Finalize: reduce 32 slice partials, combine with x, divide by separable sum.
// ---------------------------------------------------------------------------
__global__ __launch_bounds__(256) void tsk_finalize(
    const float* __restrict__ x,
    const float* __restrict__ center,
    const float* __restrict__ sigma,
    const float* __restrict__ part,
    float* __restrict__ out)
{
    const int n = blockIdx.x * blockDim.x + threadIdx.x;
    if (n >= N_SAMPLES) return;
    const int tile = n >> 4;
    const int row  = n & 15;

    float G[9];
    #pragma unroll
    for (int m = 0; m < 9; ++m) G[m] = 0.0f;
    for (int s = 0; s < SLICES; ++s) {
        const float* p = part + ((size_t)s * NTILES + tile) * 256 + row * 16;
        #pragma unroll
        for (int m = 0; m < 9; ++m) G[m] += p[m];
    }

    float num = G[8];
    #pragma unroll
    for (int m = 0; m < MDIM; ++m) num += x[n * MDIM + m] * G[m];

    float denom = 1.0f;
    #pragma unroll
    for (int m = 0; m < MDIM; ++m) {
        float ssum = 0.0f;
        #pragma unroll
        for (int d = 0; d < NMF; ++d) {
            const float diff = x[n * MDIM + m] - center[m * NMF + d];
            const float sg   = sigma[m * NMF + d];
            ssum += __expf(-diff * diff * 0.5f * sg * sg);
        }
        denom *= ssum;
    }
    out[n] = num / denom;
}

extern "C" void kernel_launch(void* const* d_in, const int* in_sizes, int n_in,
                              void* d_out, int out_size, void* d_ws, size_t ws_size,
                              hipStream_t stream) {
    (void)in_sizes; (void)n_in; (void)out_size; (void)ws_size;
    const float* x      = (const float*)d_in[0];
    const float* center = (const float*)d_in[1];
    const float* sigma  = (const float*)d_in[2];
    const float* W      = (const float*)d_in[3];
    const float* b      = (const float*)d_in[4];
    float* out = (float*)d_out;

    unsigned int* bpack = (unsigned int*)d_ws;                 // 2 MB
    float*        part  = (float*)((char*)d_ws + BPACK_U32 * 4); // next 2 MB

    tsk_pack_b<<<dim3(BPACK_U32 / 256), dim3(256), 0, stream>>>(W, b, bpack);
    tsk_wmma_main<<<dim3(NTILES * BPT), dim3(256), 0, stream>>>(
        x, center, sigma, (const v8u*)bpack, part);
    tsk_finalize<<<dim3(N_SAMPLES / 256), dim3(256), 0, stream>>>(
        x, center, sigma, part, out);
}
